// Model_18408229830691
// MI455X (gfx1250) — matrix operand order repaired and checked by reading, not compiled, not run on hardware
//
#include <hip/hip_runtime.h>
#include <math.h>

// ---------------- problem constants ----------------
#define BB    64
#define SS    4096
#define HID   64
#define MODES 16
#define LIFT  256
#define NL    4
#define STILE 64              // S positions per block
#define NTILES_S (SS/STILE)   // 64
#define NCHUNK 8              // K-split for the DFT GEMM
#define KCH   (SS/NCHUNK)     // 512

typedef _Float16 f16;
typedef __attribute__((ext_vector_type(16))) _Float16 v16h;
typedef __attribute__((ext_vector_type(8)))  _Float16 v8h;
typedef __attribute__((ext_vector_type(8)))  float    v8f;

__device__ __forceinline__ float gelu_exact(float x) {
    return 0.5f * x * (1.0f + erff(x * 0.70710678118654752440f));
}

// ---- WMMA fragment loaders (CDNA5 16x16x32 f16; contiguous k = 16*half + s in both A and B) ----
// A (16x32, MxK), row-major source with leading dim ld:
//   lane: row = rbase + (lane&15); 16 contiguous K at kb + half*16
__device__ __forceinline__ v16h frag_a(const f16* __restrict__ A, int ld,
                                       int rbase, int kb, int l15, int half) {
    const f16* p = A + (long)(rbase + l15) * ld + kb + half * 16;
    v8h lo = *(const v8h*)p;
    v8h hi = *(const v8h*)(p + 8);
    v16h r;
#pragma unroll
    for (int i = 0; i < 8; ++i) { r[i] = lo[i]; r[8 + i] = hi[i]; }
    return r;
}
// B (32x16, KxN) from a "BT" array laid out [n][k] (K contiguous per column):
//   lane: col n = nbase + (lane&15); 16 contiguous K at kb + half*16
__device__ __forceinline__ v16h frag_b(const f16* __restrict__ BT, int ldk,
                                       int nbase, int kb, int l15, int half) {
    const f16* p = BT + (long)(nbase + l15) * ldk + kb + half * 16;
    return *(const v16h*)p;
}
__device__ __forceinline__ v8f wmma_f16(v16h a, v16h b, v8f c) {
    return __builtin_amdgcn_wmma_f32_16x16x32_f16(false, a, false, b,
                                                  (short)0, c, false, false);
}

// ---------------- prep: trig bases ----------------
// F[(32,S)]   rows 0..15 = cos(2pi m s/S), rows 16..31 = -sin(...)   (forward DFT, K=s contiguous)
// GinvT[(S,32)] per s: k<16 -> (k==0?1:2)*cos ; k>=16 -> (m==0?0:-2)*sin  (1/S folded into yk)
__global__ void k_basis(f16* __restrict__ F, f16* __restrict__ GinvT) {
    int s = blockIdx.x * blockDim.x + threadIdx.x;
    if (s >= SS) return;
#pragma unroll 1
    for (int m = 0; m < MODES; ++m) {
        float th = (float)(6.283185307179586 * (double)m * (double)s / (double)SS);
        float sn, cs;
        sincosf(th, &sn, &cs);
        F[m * SS + s]            = (f16)cs;
        F[(16 + m) * SS + s]     = (f16)(-sn);
        GinvT[s * 32 + m]        = (f16)((m == 0 ? 1.0f : 2.0f) * cs);
        GinvT[s * 32 + 16 + m]   = (m == 0) ? (f16)0.0f : (f16)(-2.0f * sn);
    }
}

__global__ void k_cvt_f16(const float* __restrict__ src, f16* __restrict__ dst, int n) {
    int i = blockIdx.x * blockDim.x + threadIdx.x;
    if (i < n) dst[i] = (f16)src[i];
}

// ---------------- lift: x(B,S,2) -> h(B,*,*) ----------------
// stage1 (K=2) in VALU -> LDS f16 [s][o]; stage2 64x256 WMMA GEMM.
__global__ void __launch_bounds__(256)
k_lift(const float* __restrict__ x, const float* __restrict__ w1,
       const float* __restrict__ b1, const f16* __restrict__ w2_16,
       const float* __restrict__ b2,
       f16* __restrict__ h_sc, f16* __restrict__ h_cs) {
    __shared__ __align__(32) f16 tLds[STILE * LIFT];   // 32 KB, [s][o]
    const int blk = blockIdx.x;
    const int b   = blk / NTILES_S;
    const int s0  = (blk % NTILES_S) * STILE;
    const int tid = threadIdx.x;

    // ---- stage 1: t[o][s] = gelu(w1 @ x + b1) ----
    {
        const int o = tid;                     // 0..255
        const float a0 = w1[o * 2 + 0], a1 = w1[o * 2 + 1], bb = b1[o];
        const float* xp = x + ((long)b * SS + s0) * 2;
#pragma unroll 4
        for (int s = 0; s < STILE; ++s) {
            float v = fmaf(a0, xp[s * 2 + 0], fmaf(a1, xp[s * 2 + 1], bb));
            tLds[s * LIFT + o] = (f16)gelu_exact(v);
        }
    }
    __syncthreads();

    // ---- stage 2: h(c,s) = w2(64x256) @ t(256x64) ----
    const int w = tid >> 5, lane = tid & 31, half = lane >> 4, l15 = lane & 15;
    const int mtile = w >> 1, npair = w & 1;   // 8 waves: 4 m-tiles x 2 n-pairs

    for (int t = 0; t < 2; ++t) {
        const int ntile = npair * 2 + t;
        v8f acc;
#pragma unroll
        for (int v = 0; v < 8; ++v) acc[v] = b2[mtile * 16 + 8 * half + v];
#pragma unroll 2
        for (int kb = 0; kb < LIFT; kb += 32) {
            v16h a  = frag_a(w2_16, LIFT, mtile * 16, kb, l15, half);
            v16h bf = frag_b(tLds, LIFT, ntile * 16, kb, l15, half); // LDS reads
            acc = wmma_f16(a, bf, acc);
        }
        const int s  = s0 + ntile * 16 + l15;
        const int c0 = mtile * 16 + 8 * half;
        v8h hv;
#pragma unroll
        for (int v = 0; v < 8; ++v) hv[v] = (f16)acc[v];
        *(v8h*)(h_sc + ((long)b * SS + s) * HID + c0) = hv;
#pragma unroll
        for (int v = 0; v < 8; ++v)
            h_cs[((long)b * HID + c0 + v) * SS + s] = hv[v];
    }
}

// ---------------- DFT (K-split): xfp[ch] = h_cs(HIDx512) @ F^T chunk ----------------
__global__ void __launch_bounds__(256)
k_dft(const f16* __restrict__ h_cs, const f16* __restrict__ F,
      float* __restrict__ xfp) {
    const int blk = blockIdx.x;
    const int b   = blk >> 3;          // batch
    const int ch  = blk & (NCHUNK - 1);// K chunk
    const int tid = threadIdx.x;
    const int w = tid >> 5, lane = tid & 31, half = lane >> 4, l15 = lane & 15;
    const int mtile = w >> 1, ntile = w & 1;   // 4 m-tiles x 2 n-tiles = 8 waves

    const f16* A = h_cs + (long)b * HID * SS;  // (HID x S) row-major, ld=S
    const int k0 = ch * KCH;
    v8f acc = {};
#pragma unroll 2
    for (int kb = k0; kb < k0 + KCH; kb += 32) {
        const f16* ap = A + (long)(mtile * 16 + l15) * SS + kb;
        __builtin_prefetch(ap + 256, 0, 1);    // global_prefetch next tiles
        v16h a  = frag_a(A, SS, mtile * 16, kb, l15, half);
        v16h bf = frag_b(F, SS, ntile * 16, kb, l15, half);
        acc = wmma_f16(a, bf, acc);
    }
    const int kq = ntile * 16 + l15;           // 0..31 (Re 0..15, Im 16..31)
    const int c0 = mtile * 16 + 8 * half;
#pragma unroll
    for (int v = 0; v < 8; ++v)
        xfp[(((long)ch * BB + b) * HID + c0 + v) * 32 + kq] = acc[v];
}

// deterministic partial-sum reduction: xf = sum_ch xfp[ch]
__global__ void k_xfsum(const float* __restrict__ xfp, float* __restrict__ xf) {
    const int i = blockIdx.x * blockDim.x + threadIdx.x;
    if (i >= BB * HID * 32) return;
    float s = 0.f;
#pragma unroll
    for (int ch = 0; ch < NCHUNK; ++ch)
        s += xfp[(long)ch * BB * HID * 32 + i];
    xf[i] = s;
}

// ---------------- complex 64x64 mode mix (tiny, VALU f32) ----------------
__global__ void k_mix(const float* __restrict__ xf,
                      const float* __restrict__ wr, const float* __restrict__ wi,
                      f16* __restrict__ yk16) {
    const int id = blockIdx.x * blockDim.x + threadIdx.x;  // B*HID*MODES = 65536
    const int m = id & 15, o = (id >> 4) & 63, b = id >> 10;
    float yr = 0.f, yi = 0.f;
    const float* xb = xf + (long)b * HID * 32;
#pragma unroll 4
    for (int i = 0; i < HID; ++i) {
        float xr = xb[i * 32 + m], xi = xb[i * 32 + 16 + m];
        float ar = wr[(i * HID + o) * MODES + m];
        float ai = wi[(i * HID + o) * MODES + m];
        yr = fmaf(xr, ar, fmaf(-xi, ai, yr));
        yi = fmaf(xr, ai, fmaf( xi, ar, yi));
    }
    const float inv = 1.0f / (float)SS;        // fold irfft 1/S here
    yk16[((long)b * HID + o) * 32 + m]      = (f16)(yr * inv);
    yk16[((long)b * HID + o) * 32 + 16 + m] = (f16)(yi * inv);
}

// ---------------- fused layer: irfft GEMM + skip GEMM + bias (+gelu) ----------------
// h tile staged in LDS once (8 KB); B fragments come from LDS -> 4x less global read
// and no read/overwrite hazard on h_sc.
__global__ void __launch_bounds__(256)
k_layer(const f16* __restrict__ yk16, const f16* __restrict__ GinvT,
        const f16* __restrict__ skw16, const float* __restrict__ skb,
        f16* __restrict__ h_sc, f16* __restrict__ h_cs, int apply_gelu) {
    __shared__ __align__(32) f16 hLds[STILE * HID];   // [s][c], 8 KB
    const int blk = blockIdx.x;
    const int b   = blk / NTILES_S;
    const int s0  = (blk % NTILES_S) * STILE;
    const int tid = threadIdx.x;
    const int w = tid >> 5, lane = tid & 31, half = lane >> 4, l15 = lane & 15;
    const int mtile = w >> 1, npair = w & 1;

    // stage h tile: 256 threads x one 32B chunk = 8 KB, fully coalesced
    {
        const v16h* src = (const v16h*)(h_sc + ((long)b * SS + s0) * HID);
        ((v16h*)hLds)[tid] = src[tid];
    }
    __syncthreads();

    const f16* Ay = yk16 + (long)b * HID * 32;        // (64 x 32), ld=32

    for (int t = 0; t < 2; ++t) {
        const int ntile = npair * 2 + t;
        v8f acc;
#pragma unroll
        for (int v = 0; v < 8; ++v) acc[v] = skb[mtile * 16 + 8 * half + v];
        // spectral: yk(64x32) @ Ginv(32xS)   -- single K step
        {
            v16h a  = frag_a(Ay, 32, mtile * 16, 0, l15, half);
            v16h bf = frag_b(GinvT, 32, s0 + ntile * 16, 0, l15, half);
            acc = wmma_f16(a, bf, acc);
        }
        // skip: W(64x64) @ h(64xS)  -- two K steps, B frags from LDS tile
#pragma unroll
        for (int kb = 0; kb < HID; kb += 32) {
            v16h a  = frag_a(skw16, HID, mtile * 16, kb, l15, half);
            v16h bf = frag_b(hLds, HID, ntile * 16, kb, l15, half);
            acc = wmma_f16(a, bf, acc);
        }
        const int s  = s0 + ntile * 16 + l15;
        const int c0 = mtile * 16 + 8 * half;
        v8h hv;
#pragma unroll
        for (int v = 0; v < 8; ++v) {
            float val = acc[v];
            if (apply_gelu) val = gelu_exact(val);
            hv[v] = (f16)val;
        }
        *(v8h*)(h_sc + ((long)b * SS + s) * HID + c0) = hv;
#pragma unroll
        for (int v = 0; v < 8; ++v)
            h_cs[((long)b * HID + c0 + v) * SS + s] = hv[v];
    }
}

// ---------------- projection + partial mean ----------------
__global__ void __launch_bounds__(256)
k_proj(const f16* __restrict__ h_sc, const f16* __restrict__ pw1_16,
       const float* __restrict__ pb1, const float* __restrict__ pw2,
       float* __restrict__ partials) {
    __shared__ __align__(32) f16 hLds[STILE * HID];    // [s][c], 8 KB
    __shared__ __align__(32) f16 gLds[STILE * LIFT];   // [s][o], 32 KB
    __shared__ float redLds[3 * STILE];
    const int blk = blockIdx.x;
    const int b   = blk / NTILES_S;
    const int st  = blk % NTILES_S;
    const int s0  = st * STILE;
    const int tid = threadIdx.x;
    const int w = tid >> 5, lane = tid & 31, half = lane >> 4, l15 = lane & 15;

    // stage h tile once (was read 16x from global before)
    {
        const v16h* src = (const v16h*)(h_sc + ((long)b * SS + s0) * HID);
        ((v16h*)hLds)[tid] = src[tid];
    }
    __syncthreads();

    // g(256 x 64) = gelu(pw1(256x64) @ h(64x64) + b1); 64 subtiles / 8 waves
    for (int i = 0; i < 8; ++i) {
        const int sub = w * 8 + i;
        const int mtile = sub >> 2, ntile = sub & 3;
        v8f acc;
#pragma unroll
        for (int v = 0; v < 8; ++v) acc[v] = pb1[mtile * 16 + 8 * half + v];
#pragma unroll
        for (int kb = 0; kb < HID; kb += 32) {
            v16h a  = frag_a(pw1_16, HID, mtile * 16, kb, l15, half);
            v16h bf = frag_b(hLds, HID, ntile * 16, kb, l15, half);
            acc = wmma_f16(a, bf, acc);
        }
        const int s  = ntile * 16 + l15;
        const int o0 = mtile * 16 + 8 * half;
        v8h gv;
#pragma unroll
        for (int v = 0; v < 8; ++v) gv[v] = (f16)gelu_exact(acc[v]);
        *(v8h*)(gLds + s * LIFT + o0) = gv;                  // ds_store_b128
    }
    __syncthreads();

    // out3[s] = pw2(3x256) @ g ; reduce over s within tile
    if (tid < 3 * STILE) {
        const int o3 = tid / STILE, s = tid % STILE;
        float sum = 0.f;
        const float* wrow = pw2 + o3 * LIFT;
#pragma unroll 4
        for (int o = 0; o < LIFT; ++o)
            sum = fmaf(wrow[o], (float)gLds[s * LIFT + o], sum);
        redLds[o3 * STILE + s] = sum;
    }
    __syncthreads();
    if (tid < 3) {
        float tot = 0.f;
        for (int s = 0; s < STILE; ++s) tot += redLds[tid * STILE + s];
        partials[((long)b * NTILES_S + st) * 3 + tid] = tot;
    }
}

__global__ void k_reduce(const float* __restrict__ partials,
                         const float* __restrict__ pb2, float* __restrict__ out) {
    const int t = blockIdx.x * blockDim.x + threadIdx.x;
    if (t >= BB * 3) return;
    const int b = t / 3, o3 = t % 3;
    float sum = 0.f;
    for (int tile = 0; tile < NTILES_S; ++tile)
        sum += partials[((long)b * NTILES_S + tile) * 3 + o3];
    out[b * 3 + o3] = pb2[o3] + sum / (float)SS;
}

// ---------------- host orchestration ----------------
extern "C" void kernel_launch(void* const* d_in, const int* in_sizes, int n_in,
                              void* d_out, int out_size, void* d_ws, size_t ws_size,
                              hipStream_t stream) {
    (void)in_sizes; (void)n_in; (void)out_size; (void)ws_size;
    const float* x       = (const float*)d_in[0];
    const float* lift_w1 = (const float*)d_in[1];
    const float* lift_b1 = (const float*)d_in[2];
    const float* lift_w2 = (const float*)d_in[3];
    const float* lift_b2 = (const float*)d_in[4];
    const float* spec_wr = (const float*)d_in[5];
    const float* spec_wi = (const float*)d_in[6];
    const float* skip_w  = (const float*)d_in[7];
    const float* skip_b  = (const float*)d_in[8];
    const float* proj_w1 = (const float*)d_in[9];
    const float* proj_b1 = (const float*)d_in[10];
    const float* proj_w2 = (const float*)d_in[11];
    const float* proj_b2 = (const float*)d_in[12];
    float* out = (float*)d_out;

    char* ws = (char*)d_ws;
    size_t off = 0;
    auto alloc = [&](size_t bytes) { char* p = ws + off; off = (off + bytes + 255) & ~(size_t)255; return p; };
    f16*   h_sc   = (f16*)  alloc((size_t)BB * SS * HID * 2);
    f16*   h_cs   = (f16*)  alloc((size_t)BB * SS * HID * 2);
    f16*   F      = (f16*)  alloc((size_t)32 * SS * 2);
    f16*   GinvT  = (f16*)  alloc((size_t)SS * 32 * 2);
    float* xfp    = (float*)alloc((size_t)NCHUNK * BB * HID * 32 * 4);
    float* xf     = (float*)alloc((size_t)BB * HID * 32 * 4);
    f16*   yk16   = (f16*)  alloc((size_t)BB * HID * 32 * 2);
    f16*   w2_16  = (f16*)  alloc((size_t)HID * LIFT * 2);
    f16*   skw16  = (f16*)  alloc((size_t)NL * HID * HID * 2);
    f16*   pw1_16 = (f16*)  alloc((size_t)LIFT * HID * 2);
    float* parts  = (float*)alloc((size_t)BB * NTILES_S * 3 * 4);

    // prep
    k_basis<<<SS / 256, 256, 0, stream>>>(F, GinvT);
    k_cvt_f16<<<(HID * LIFT + 255) / 256, 256, 0, stream>>>(lift_w2, w2_16, HID * LIFT);
    k_cvt_f16<<<(NL * HID * HID + 255) / 256, 256, 0, stream>>>(skip_w, skw16, NL * HID * HID);
    k_cvt_f16<<<(LIFT * HID + 255) / 256, 256, 0, stream>>>(proj_w1, pw1_16, LIFT * HID);

    const int nblk = BB * NTILES_S;  // 4096
    k_lift<<<nblk, 256, 0, stream>>>(x, lift_w1, lift_b1, w2_16, lift_b2, h_sc, h_cs);

    for (int l = 0; l < NL; ++l) {
        k_dft<<<BB * NCHUNK, 256, 0, stream>>>(h_cs, F, xfp);
        k_xfsum<<<(BB * HID * 32 + 255) / 256, 256, 0, stream>>>(xfp, xf);
        k_mix<<<(BB * HID * MODES) / 256, 256, 0, stream>>>(
            xf, spec_wr + (long)l * HID * HID * MODES,
            spec_wi + (long)l * HID * HID * MODES, yk16);
        k_layer<<<nblk, 256, 0, stream>>>(yk16, GinvT, skw16 + (long)l * HID * HID,
                                          skip_b + (long)l * HID, h_sc, h_cs,
                                          l < NL - 1 ? 1 : 0);
    }

    k_proj<<<nblk, 256, 0, stream>>>(h_sc, pw1_16, proj_b1, proj_w2, parts);
    k_reduce<<<1, 256, 0, stream>>>(parts, proj_b2, out);
}
